// GCN3D_Mar28_PoolingDeepGlobal_32461362823181
// MI455X (gfx1250) — matrix operand order repaired and checked by reading, not compiled, not run on hardware
//
#include <hip/hip_runtime.h>
#include <hip/hip_bf16.h>

typedef __attribute__((ext_vector_type(16))) __bf16 v16bf;
typedef __attribute__((ext_vector_type(8)))  float  v8f;
typedef __attribute__((ext_vector_type(4)))  float  v4f;

#define WAVE 32
#define COL_CHUNK 4   // column tiles per wave in the WMMA GEMM

__device__ __forceinline__ float elu1(float v) {
    return v > 0.f ? v : (__expf(v) - 1.f);
}

// ---------------------------------------------------------------------------
// WMMA GEMM: Y[M,O] = act( X[M,K] @ W[O,K]^T (+ bias) )
// M%16==0, O%16==0, K = KT*32 (template). One wave per (16-row strip ×
// COL_CHUNK column tiles); the full-K A fragments live in registers
// (KT<=8 -> 64 VGPRs) so A is loaded once per strip.
// ---------------------------------------------------------------------------
template<int KT, bool BIAS_ELU>
__global__ void gemm_wmma(const float* __restrict__ X,
                          const float* __restrict__ W,
                          const float* __restrict__ bias,
                          float* __restrict__ Y,
                          int M, int O) {
    constexpr int K = KT * 32;
    const int wavesPerBlock = blockDim.x / WAVE;
    const int waveId = blockIdx.x * wavesPerBlock + (threadIdx.x / WAVE);
    const int colTiles = O >> 4;
    const int nChunks  = (colTiles + COL_CHUNK - 1) / COL_CHUNK;
    const int rowTiles = M >> 4;
    if (waveId >= rowTiles * nChunks) return;            // wave-uniform exit
    const int rowTile = waveId / nChunks;
    const int chunk   = waveId % nChunks;

    const int lane = threadIdx.x & (WAVE - 1);
    const int l15  = lane & 15;
    const int hi   = lane >> 4;                          // K-half selector

    // ---- load A fragments for the whole K extent (ISA 16-bit A layout) ----
    const float* xrow = X + (size_t)(rowTile * 16 + l15) * K;
    v16bf afrag[KT];
#pragma unroll
    for (int kt = 0; kt < KT; ++kt) {
        const int ka = kt * 32 + hi * 8;                 // {ka..ka+7, ka+16..ka+23}
#pragma unroll
        for (int j = 0; j < 8; ++j) {
            afrag[kt][j]     = (__bf16)xrow[ka + j];
            afrag[kt][8 + j] = (__bf16)xrow[ka + 16 + j];
        }
    }

    const int ctEnd = min(colTiles, (chunk + 1) * COL_CHUNK);
    for (int ct = chunk * COL_CHUNK; ct < ctEnd; ++ct) {
        const int col = ct * 16 + l15;
        const float* wrow = W + (size_t)col * K;         // B^T column = W row
        v8f acc = {};
#pragma unroll
        for (int kt = 0; kt < KT; ++kt) {
            v16bf b;
            const int kb = kt * 32 + hi * 8;             // {kb..kb+7, kb+16..kb+23}
#pragma unroll
            for (int j = 0; j < 16; ++j) b[j] = (__bf16)wrow[kb + j + (j >= 8 ? 8 : 0)];
            acc = __builtin_amdgcn_wmma_f32_16x16x32_bf16(
                false, afrag[kt], false, b, (short)0, acc, false, false);
        }
        float bv = 0.f;
        if (BIAS_ELU) bv = bias[col];
#pragma unroll
        for (int r = 0; r < 8; ++r) {
            const int row = rowTile * 16 + hi * 8 + r;
            float v = acc[r] + bv;
            if (BIAS_ELU) v = elu1(v);
            Y[(size_t)row * O + col] = v;
        }
    }
}

// ---------------------------------------------------------------------------
// Tiny scalar GEMM for the last coarse layers (K or O not WMMA-shaped).
// ---------------------------------------------------------------------------
template<bool BIAS_ELU>
__global__ void gemm_small(const float* __restrict__ X, const float* __restrict__ W,
                           const float* __restrict__ bias, float* __restrict__ Y,
                           int M, int K, int O) {
    int idx = blockIdx.x * blockDim.x + threadIdx.x;
    if (idx >= M * O) return;
    int m = idx / O, o = idx % O;
    float s = BIAS_ELU ? bias[o] : 0.f;
    const float* xr = X + (size_t)m * K;
    const float* wr = W + (size_t)o * K;
    for (int k = 0; k < K; ++k) s += xr[k] * wr[k];
    Y[idx] = BIAS_ELU ? elu1(s) : s;
}

// ---------------------------------------------------------------------------
// Degree helpers (deg includes self-loop, then in-place rsqrt -> dinv)
// ---------------------------------------------------------------------------
__global__ void fill_ones(float* p, int n) {
    int i = blockIdx.x * blockDim.x + threadIdx.x;
    if (i < n) p[i] = 1.0f;
}
__global__ void deg_add(const int* __restrict__ dst, float* deg, int nE) {
    int e = blockIdx.x * blockDim.x + threadIdx.x;
    if (e < nE) atomicAdd(&deg[dst[e]], 1.0f);
}
__global__ void deg_rsqrt(float* p, int n) {
    int i = blockIdx.x * blockDim.x + threadIdx.x;
    if (i < n) p[i] = rsqrtf(fmaxf(p[i], 1.0f));
}

// ---------------------------------------------------------------------------
// Edge scatter: agg[dst] += dinv[src]*dinv[dst] * h[src]   (per feature)
// ---------------------------------------------------------------------------
__global__ void scatter_edges(const int* __restrict__ src, const int* __restrict__ dst,
                              const float* __restrict__ dinv, const float* __restrict__ h,
                              float* __restrict__ agg, int nE, int F) {
    long long total = (long long)nE * F;
    long long stride = (long long)gridDim.x * blockDim.x;
    for (long long i = blockIdx.x * (long long)blockDim.x + threadIdx.x;
         i < total; i += stride) {
        int e = (int)(i / F);
        int f = (int)(i - (long long)e * F);
        int s = src[e], d = dst[e];
        float w = dinv[s] * dinv[d];
        atomicAdd(&agg[(size_t)d * F + f], w * h[(size_t)s * F + f]);
    }
}

// out = act( agg + dinv[i]^2 * h[i] + b )   (self-loop + bias + optional elu)
template<bool ELU>
__global__ void selfloop_bias_act(const float* __restrict__ h, const float* __restrict__ dinv,
                                  const float* __restrict__ agg, const float* __restrict__ b,
                                  float* __restrict__ out, int n, int F) {
    long long total = (long long)n * F;
    long long i = blockIdx.x * (long long)blockDim.x + threadIdx.x;
    if (i >= total) return;
    int node = (int)(i / F);
    int f = (int)(i - (long long)node * F);
    float d = dinv[node];
    float v = agg[i] + d * d * h[i] + b[f];
    out[i] = ELU ? elu1(v) : v;
}

// ---------------------------------------------------------------------------
// InstanceNorm stats: per-channel mean / rsqrt(var+eps) over nodes
// ---------------------------------------------------------------------------
__global__ void inorm_stats(const float* __restrict__ x, float* mu, float* rinv,
                            int n, int F) {
    int c = blockIdx.x;
    float s = 0.f, q = 0.f;
    for (int i = threadIdx.x; i < n; i += blockDim.x) {
        float v = x[(size_t)i * F + c];
        s += v; q += v * v;
    }
    __shared__ float ss[256], sq[256];
    ss[threadIdx.x] = s; sq[threadIdx.x] = q;
    __syncthreads();
    for (int off = blockDim.x >> 1; off > 0; off >>= 1) {
        if ((int)threadIdx.x < off) {
            ss[threadIdx.x] += ss[threadIdx.x + off];
            sq[threadIdx.x] += sq[threadIdx.x + off];
        }
        __syncthreads();
    }
    if (threadIdx.x == 0) {
        float m = ss[0] / n;
        float var = sq[0] / n - m * m;
        mu[c] = m;
        rinv[c] = rsqrtf(var + 1e-5f);
    }
}

// cluster count and normalized-feature pooling accumulation
__global__ void cnt_accum(const int* __restrict__ cluster, float* cnt, int n) {
    int i = blockIdx.x * blockDim.x + threadIdx.x;
    if (i < n) atomicAdd(&cnt[cluster[i]], 1.0f);
}
__global__ void pool_accum(const float* __restrict__ x, const float* __restrict__ mu,
                           const float* __restrict__ rinv, const int* __restrict__ cluster,
                           float* __restrict__ cx, int n, int F) {
    long long total = (long long)n * F;
    long long i = blockIdx.x * (long long)blockDim.x + threadIdx.x;
    if (i >= total) return;
    int node = (int)(i / F);
    int f = (int)(i - (long long)node * F);
    float y = (x[i] - mu[f]) * rinv[f];
    atomicAdd(&cx[(size_t)cluster[node] * F + f], y);
}
__global__ void pool_div(float* cx, const float* __restrict__ cnt, int Cn, int F) {
    int i = blockIdx.x * blockDim.x + threadIdx.x;
    if (i >= Cn * F) return;
    cx[i] /= fmaxf(cnt[i / F], 1.0f);
}

// ---------------------------------------------------------------------------
// o_feat = W_fcO[Rows,Kd] @ g_feat[Kd]; bandwidth bound (614 MB of W read
// exactly once -> non-temporal loads so it does not evict L2-resident
// activations). One wave per row, g_feat staged in LDS.
// ---------------------------------------------------------------------------
__global__ void gemv_fco(const float* __restrict__ W, const float* __restrict__ g,
                         float* __restrict__ o, int Rows, int Kd) {
    __shared__ float gs[2560];
    for (int i = threadIdx.x; i < Kd; i += blockDim.x) gs[i] = g[i];
    __syncthreads();
    const int wavesPerBlock = blockDim.x / WAVE;
    const int row = blockIdx.x * wavesPerBlock + threadIdx.x / WAVE;
    if (row >= Rows) return;
    const int lane = threadIdx.x & (WAVE - 1);
    const float* wr = W + (size_t)row * Kd;
    float s = 0.f;
    for (int k = lane * 4; k < Kd; k += WAVE * 4) {
        __builtin_prefetch(wr + k + WAVE * 8, 0, 1);        // global_prefetch_b8
        v4f w4 = __builtin_nontemporal_load((const v4f*)(wr + k));  // TH=NT stream
        s += w4.x * gs[k] + w4.y * gs[k + 1] + w4.z * gs[k + 2] + w4.w * gs[k + 3];
    }
    for (int off = 16; off > 0; off >>= 1) s += __shfl_down(s, off, WAVE);
    if (lane == 0) o[row] = s;
}

// ---------------------------------------------------------------------------
// Host-side launcher
// ---------------------------------------------------------------------------
static inline int ceil_div(long long a, int b) { return (int)((a + b - 1) / b); }

static void launch_gemm(const float* X, const float* W, const float* bias, float* Y,
                        int M, int K, int O, bool biasElu, hipStream_t s) {
    const int colTiles = O / 16;
    const int nChunks  = (colTiles + COL_CHUNK - 1) / COL_CHUNK;
    const int waves    = (M / 16) * nChunks;
    const int blocks   = (waves + 7) / 8;                 // 8 waves / 256-thread block
#define DISPATCH_KT(KT)                                                          \
    do {                                                                         \
        if (biasElu) gemm_wmma<KT, true ><<<blocks, 256, 0, s>>>(X, W, bias, Y, M, O); \
        else         gemm_wmma<KT, false><<<blocks, 256, 0, s>>>(X, W, bias, Y, M, O); \
    } while (0)
    switch (K / 32) {
        case 1: DISPATCH_KT(1); break;
        case 2: DISPATCH_KT(2); break;
        case 3: DISPATCH_KT(3); break;
        case 4: DISPATCH_KT(4); break;
        case 8: DISPATCH_KT(8); break;
        default: break;                                   // not used by this model
    }
#undef DISPATCH_KT
}

static void launch_scatter(const int* src, const int* dst, const float* dinv,
                           const float* h, float* agg, int nE, int F, hipStream_t s) {
    long long total = (long long)nE * F;
    int blocks = ceil_div(total, 256);
    if (blocks > (1 << 20)) blocks = 1 << 20;
    scatter_edges<<<blocks, 256, 0, s>>>(src, dst, dinv, h, agg, nE, F);
}

extern "C" void kernel_launch(void* const* d_in, const int* in_sizes, int n_in,
                              void* d_out, int out_size, void* d_ws, size_t ws_size,
                              hipStream_t stream) {
    // ---- inputs (setup_inputs order) ----
    const float* x     = (const float*)d_in[0];
    const float* W_G1  = (const float*)d_in[1];  const float* b_G1  = (const float*)d_in[2];
    const float* Wf_G1 = (const float*)d_in[3];  const float* bf_G1 = (const float*)d_in[4];
    const float* W_G2  = (const float*)d_in[5];  const float* b_G2  = (const float*)d_in[6];
    const float* Wf_G2 = (const float*)d_in[7];  const float* bf_G2 = (const float*)d_in[8];
    const float* W_L1  = (const float*)d_in[9];  const float* b_L1  = (const float*)d_in[10];
    const float* Wf_L1 = (const float*)d_in[11]; const float* bf_L1 = (const float*)d_in[12];
    const float* W_L2  = (const float*)d_in[13]; const float* b_L2  = (const float*)d_in[14];
    const float* Wf_L2 = (const float*)d_in[15]; const float* bf_L2 = (const float*)d_in[16];
    const float* W_M1  = (const float*)d_in[17]; const float* b_M1  = (const float*)d_in[18];
    const float* Wf_M1 = (const float*)d_in[19]; const float* bf_M1 = (const float*)d_in[20];
    const float* W_O1  = (const float*)d_in[21]; const float* b_O1  = (const float*)d_in[22];
    const float* W_fcO = (const float*)d_in[23];
    const int* edge    = (const int*)d_in[24];
    const int* cluster = (const int*)d_in[25];
    const int* cedge   = (const int*)d_in[27];

    const int N  = in_sizes[25];                 // 20000
    const int E  = in_sizes[24] / 2;             // 640000
    const int EC = in_sizes[27] / 2;             // coarse edge count (dynamic)
    const int C  = in_sizes[23] / (3 * N * 5);   // 512

    const int* srcF = edge;      const int* dstF = edge + E;
    const int* srcC = cedge;     const int* dstC = cedge + EC;

    // ---- workspace layout (floats) ----
    float* ws = (float*)d_ws;
    size_t o = 0;
    float* bufA  = ws + o; o += (size_t)N * 256;
    float* bufB  = ws + o; o += (size_t)N * 256;
    float* bufC  = ws + o; o += (size_t)N * 256;
    float* dinvF = ws + o; o += N;
    float* mu    = ws + o; o += 256;
    float* rinv  = ws + o; o += 256;
    float* cx    = ws + o; o += (size_t)C * 256;
    float* cnt   = ws + o; o += C;
    float* za    = ws + o; o += (size_t)C * 256;
    float* zb    = ws + o; o += (size_t)C * 256;
    float* zc    = ws + o; o += (size_t)C * 256;
    float* dinvC = ws + o; o += C;
    (void)ws_size; (void)n_in; (void)out_size;

    float* g_feat = (float*)d_out;               // [5*C]
    float* o_feat = g_feat + 5 * C;              // [3*N]

    // ---- fine-graph degree -> dinv ----
    fill_ones<<<ceil_div(N, 256), 256, 0, stream>>>(dinvF, N);
    deg_add<<<ceil_div(E, 256), 256, 0, stream>>>(dstF, dinvF, E);
    deg_rsqrt<<<ceil_div(N, 256), 256, 0, stream>>>(dinvF, N);

    // GCN G1: h = x @ W_G1^T  [N,64]
    launch_gemm(x, W_G1, nullptr, bufB, N, 128, 64, false, stream);
    hipMemsetAsync(bufC, 0, (size_t)N * 64 * sizeof(float), stream);
    launch_scatter(srcF, dstF, dinvF, bufB, bufC, E, 64, stream);
    selfloop_bias_act<true><<<ceil_div((long long)N * 64, 256), 256, 0, stream>>>(
        bufB, dinvF, bufC, b_G1, bufA, N, 64);
    // FC G1: [N,64] -> [N,128]
    launch_gemm(bufA, Wf_G1, bf_G1, bufB, N, 64, 128, true, stream);
    // GCN G2: h = x @ W_G2^T [N,256]
    launch_gemm(bufB, W_G2, nullptr, bufA, N, 128, 256, false, stream);
    hipMemsetAsync(bufC, 0, (size_t)N * 256 * sizeof(float), stream);
    launch_scatter(srcF, dstF, dinvF, bufA, bufC, E, 256, stream);
    selfloop_bias_act<true><<<ceil_div((long long)N * 256, 256), 256, 0, stream>>>(
        bufA, dinvF, bufC, b_G2, bufB, N, 256);
    // FC G2: [N,256] -> [N,256]
    launch_gemm(bufB, Wf_G2, bf_G2, bufA, N, 256, 256, true, stream);

    // InstanceNorm(256) + avg_pool over clusters -> cx [C,256]
    inorm_stats<<<256, 256, 0, stream>>>(bufA, mu, rinv, N, 256);
    hipMemsetAsync(cx, 0, (size_t)C * 256 * sizeof(float), stream);
    hipMemsetAsync(cnt, 0, (size_t)C * sizeof(float), stream);
    cnt_accum<<<ceil_div(N, 256), 256, 0, stream>>>(cluster, cnt, N);
    pool_accum<<<ceil_div((long long)N * 256, 256), 256, 0, stream>>>(
        bufA, mu, rinv, cluster, cx, N, 256);
    pool_div<<<ceil_div((long long)C * 256, 256), 256, 0, stream>>>(cx, cnt, C, 256);

    // ---- coarse-graph degree -> dinv ----
    fill_ones<<<ceil_div(C, 256), 256, 0, stream>>>(dinvC, C);
    deg_add<<<ceil_div(EC, 256), 256, 0, stream>>>(dstC, dinvC, EC);
    deg_rsqrt<<<ceil_div(C, 256), 256, 0, stream>>>(dinvC, C);

    // GCN L1: [C,256] -> [C,128]
    launch_gemm(cx, W_L1, nullptr, zb, C, 256, 128, false, stream);
    hipMemsetAsync(zc, 0, (size_t)C * 128 * sizeof(float), stream);
    launch_scatter(srcC, dstC, dinvC, zb, zc, EC, 128, stream);
    selfloop_bias_act<true><<<ceil_div((long long)C * 128, 256), 256, 0, stream>>>(
        zb, dinvC, zc, b_L1, za, C, 128);
    // FC L1: [C,128] -> [C,96]
    launch_gemm(za, Wf_L1, bf_L1, zb, C, 128, 96, true, stream);
    // GCN L2: [C,96] -> [C,64]
    launch_gemm(zb, W_L2, nullptr, za, C, 96, 64, false, stream);
    hipMemsetAsync(zc, 0, (size_t)C * 64 * sizeof(float), stream);
    launch_scatter(srcC, dstC, dinvC, za, zc, EC, 64, stream);
    selfloop_bias_act<true><<<ceil_div((long long)C * 64, 256), 256, 0, stream>>>(
        za, dinvC, zc, b_L2, zb, C, 64);
    // FC L2: [C,64] -> [C,32]
    launch_gemm(zb, Wf_L2, bf_L2, za, C, 64, 32, true, stream);
    // GCN M1: [C,32] -> [C,16]
    launch_gemm(za, W_M1, nullptr, zb, C, 32, 16, false, stream);
    hipMemsetAsync(zc, 0, (size_t)C * 16 * sizeof(float), stream);
    launch_scatter(srcC, dstC, dinvC, zb, zc, EC, 16, stream);
    selfloop_bias_act<true><<<ceil_div((long long)C * 16, 256), 256, 0, stream>>>(
        zb, dinvC, zc, b_M1, za, C, 16);
    // FC M1: [C,16] -> [C,8]  (too small for WMMA tiles)
    gemm_small<true><<<ceil_div(C * 8, 256), 256, 0, stream>>>(za, Wf_M1, bf_M1, zb, C, 16, 8);
    // GCN O1: [C,8] -> [C,5], no elu, g_feat into d_out
    gemm_small<false><<<ceil_div(C * 5, 256), 256, 0, stream>>>(zb, W_O1, nullptr, za, C, 8, 5);
    hipMemsetAsync(zc, 0, (size_t)C * 5 * sizeof(float), stream);
    launch_scatter(srcC, dstC, dinvC, za, zc, EC, 5, stream);
    selfloop_bias_act<false><<<ceil_div((long long)C * 5, 256), 256, 0, stream>>>(
        za, dinvC, zc, b_O1, g_feat, C, 5);

    // o_feat = W_fcO @ g_feat   [3N] — bandwidth-bound GEMV, NT streaming
    {
        int rows = 3 * N;
        int blocks = (rows + 7) / 8;                      // 8 waves per block
        gemv_fco<<<blocks, 256, 0, stream>>>(W_fcO, g_feat, o_feat, rows, 5 * C);
    }
}
